// DSS_82420422410646
// MI455X (gfx1250) — matrix operand order repaired and checked by reading, not compiled, not run on hardware
//
#include <hip/hip_runtime.h>
#include <math.h>

// DSS layer for MI455X (gfx1250, wave32).
//
// Pass 0 (convert_w_kernel): out_w f32 -> f16 once (1.2 MB), so the GEMM's
//   A-fragment loads are half the bytes and need zero in-loop v_cvt.
//
// Pass 1 (dss_scan_kernel): the rFFT convolution of a sum-of-64-exponentials
//   kernel == diagonal recurrence x[l] = exp(dtl)*x[l-1] + u[l],
//   y[l] = Re(sum_n c_n x_n[l]) + D*u[l], then exact-erf GELU. One wave per
//   (b,h); 2 complex states/lane packed as float2 -> v_pk_fma_f32; per-step
//   5-shuffle butterfly reduce; u broadcast via wave-uniform (readfirstlane)
//   scalar loads (SMEM path, keeps the DS pipe for the butterfly).
//   Stores g transposed gT[b][l][u] (f16, u-contiguous) for the WMMA GEMM.
//
// Pass 2 (dss_out_gemm_kernel): Out[b,v,l] = sum_u out_w[v,u]*g[b,u,l]+out_b[v]
//   = (768x768)x(768x16384) GEMM, 19.3 GFLOP. Each wave owns a 64(v)x32(l)
//   macro-tile: per K-step 4 A-frags + 2 B-frags -> 8 v_wmma_f32_16x16x32_f16,
//   0.75 KB L2 traffic per wmma. gT (25 MB) is L2-resident (192 MB L2), so the
//   GEMM runs at matrix-pipe rate; HBM cost is ~80 MB (~3.5 us @ 23.3 TB/s).

#define B_  8
#define H_  768
#define N_  64
#define L_  2048

typedef __attribute__((ext_vector_type(16))) _Float16 v16h;
typedef __attribute__((ext_vector_type(8)))  _Float16 v8h;
typedef __attribute__((ext_vector_type(4)))  _Float16 v4h;
typedef __attribute__((ext_vector_type(8)))  float    v8f;
typedef __attribute__((ext_vector_type(4)))  float    f32x4;
typedef __attribute__((ext_vector_type(2)))  float    f32x2;

__device__ __forceinline__ float gelu_erf(float x) {
    return 0.5f * x * (1.0f + erff(x * 0.70710678118654752f));
}

// ---------------------------------------------------------------------------
// Pass 0: out_w (768x768 f32) -> Wh (f16), 4 elements per thread
// ---------------------------------------------------------------------------
__global__ __launch_bounds__(256) void convert_w_kernel(
    const float* __restrict__ w, _Float16* __restrict__ wh)
{
    const int i = (blockIdx.x * 256 + threadIdx.x) * 4;
    const f32x4 x = *reinterpret_cast<const f32x4*>(w + i);
    v4h o;
    o[0] = (_Float16)x[0]; o[1] = (_Float16)x[1];
    o[2] = (_Float16)x[2]; o[3] = (_Float16)x[3];
    *reinterpret_cast<v4h*>(wh + i) = o;
}

// ---------------------------------------------------------------------------
// Pass 1: diagonal SSM scan + skip + GELU, writes gT[b][l][u] (f16, u-contig)
// ---------------------------------------------------------------------------
__global__ __launch_bounds__(256) void dss_scan_kernel(
    const float* __restrict__ u, const float* __restrict__ log_dt,
    const float* __restrict__ Lambda, const float* __restrict__ W,
    const float* __restrict__ D, _Float16* __restrict__ gT)
{
    const int lane = threadIdx.x & 31;
    int wid = blockIdx.x * (blockDim.x >> 5) + (threadIdx.x >> 5);
    wid = __builtin_amdgcn_readfirstlane(wid);   // wave-uniform for SMEM paths
    const int h = wid % H_;
    const int b = wid / H_;

    const float dt0 = expf(log_dt[2 * h + 0]);   // dt for real part
    const float dt1 = expf(log_dt[2 * h + 1]);   // dt for imag part

    // Per-lane coefficients for states n = lane and n = lane+32, packed as
    // float2 so the recurrence runs on v_pk_fma_f32/v_pk_mul_f32 (VOP3P).
    // Init 'hippo_skew_pos_imag': lam.real = -0.5 < 0 => gt0 == 0, P_max == 0,
    // dtl_neg == dtl, so kernel k[l] = Re(sum_n c_n * exp(dtl)^l).
    f32x2 vbr, vbi, vcr, vci;
    #pragma unroll
    for (int s = 0; s < 2; ++s) {
        const int n = lane + 32 * s;
        const float lre = Lambda[2 * n + 0];
        const float lim = Lambda[2 * n + 1];
        const float ar = dt0 * lre;              // Re(dtl)
        const float ai = dt1 * lim;              // Im(dtl)
        // lambda_bar = exp(dtl)
        const float ea = expf(ar);
        float sa, ca; sincosf(ai, &sa, &ca);
        const float lbr = ea * ca;
        const float lbi = ea * sa;
        // num = exp(dtl) - 1
        const float nr = lbr - 1.0f, ni = lbi;
        // den = exp(dtl * L) - 1
        const float eaL = expf(ar * (float)L_);
        float saL, caL; sincosf(ai * (float)L_, &saL, &caL);
        const float dr = eaL * caL - 1.0f, di = eaL * saL;
        // x = den * lam ; recip = conj(x) / (|x|^2 + eps)
        const float xr  = dr * lre - di * lim;
        const float xi  = dr * lim + di * lre;
        const float inv = 1.0f / (xr * xr + xi * xi + 1e-7f);
        const float rr  =  xr * inv;
        const float ri  = -xi * inv;
        // c = W * num * recip
        const float wr = W[(h * N_ + n) * 2 + 0];
        const float wi = W[(h * N_ + n) * 2 + 1];
        const float tr = wr * nr - wi * ni;
        const float ti = wr * ni + wi * nr;
        vbr[s] = lbr;            vbi[s] = lbi;
        vcr[s] = tr * rr - ti * ri;
        vci[s] = tr * ri + ti * rr;
    }
    const float Dh = D[h];
    const float* __restrict__ up = u + ((size_t)b * H_ + h) * L_;  // uniform ptr

    f32x2 xr = {0.f, 0.f}, xi = {0.f, 0.f};
    for (int base = 0; base < L_; base += 32) {
        const float ureg = up[base + lane];      // coalesced (skip connection)
        float keep = 0.0f;
        #pragma unroll
        for (int j = 0; j < 32; ++j) {
            const float uj = up[base + j];       // wave-uniform -> s_load
            const f32x2 uj2 = {uj, uj};
            // x = lambda_bar * x + u  (two complex states, packed f32)
            const f32x2 nxr = vbr * xr - vbi * xi + uj2;
            const f32x2 nxi = vbr * xi + vbi * xr;
            xr = nxr; xi = nxi;
            // Re(c * x) partial over this lane's 2 states
            const f32x2 p2 = vcr * xr - vci * xi;
            float part = p2[0] + p2[1];
            // butterfly sum over all 32 lanes (64 states)
            #pragma unroll
            for (int off = 16; off > 0; off >>= 1)
                part += __shfl_xor(part, off, 32);
            if (lane == j) keep = part;          // lane j keeps y[base+j]
        }
        const float y = keep + Dh * ureg;        // skip: l = base+lane
        const float g = gelu_erf(y);
        const int l = base + lane;
        gT[((size_t)b * L_ + l) * H_ + h] = (_Float16)g;
    }
}

// ---------------------------------------------------------------------------
// Pass 2: Out[b,v,l] = sum_u Wh[v,u] * g[b,u,l] + out_b[v]  via WMMA f16.
// One wave owns a 64(v) x 32(l) macro-tile = 4x2 wmma tiles, K=768 in steps
// of 32 -> 8 wmma per K-step. Fragment layouts per CDNA5 ISA:
//   A (16x32 f16): lane q(,+16): elems 0-7 -> K=8*g2+e, elems 8-15 -> K=16+8*g2+e
//   B (32x16 f16): lane q(,+16): elems 0-7 -> K=8*g2+e, elems 8-15 -> K=16+8*g2+e
//   C/D (16x16 f32): VGPR r -> M = r + 8*g2, N = q
// ---------------------------------------------------------------------------
__global__ __launch_bounds__(256) void dss_out_gemm_kernel(
    const _Float16* __restrict__ gT, const _Float16* __restrict__ Wh,
    const float* __restrict__ out_b, float* __restrict__ Out)
{
    const int lane = threadIdx.x & 31;
    int wid = blockIdx.x * 8 + (threadIdx.x >> 5);
    wid = __builtin_amdgcn_readfirstlane(wid);
    const int lb = wid & 63;                 // 64 l-blocks of 32
    const int t2 = wid >> 6;
    const int vb = t2 % (H_ / 64);           // 12 v-blocks of 64
    const int b  = t2 / (H_ / 64);
    const int vBase = vb * 64, lBase = lb * 32;
    const int g2 = lane >> 4, q = lane & 15;

    // Per-lane fragment base pointers (A rows already offset by 8*g2,
    // B cols already offset by 8*g2 per the ISA layout).
    const _Float16* __restrict__ aRow[4];
    #pragma unroll
    for (int i = 0; i < 4; ++i)
        aRow[i] = Wh + (size_t)(vBase + 16 * i + q) * H_ + 8 * g2;
    const _Float16* __restrict__ bCol[2];
    #pragma unroll
    for (int j = 0; j < 2; ++j)
        bCol[j] = gT + ((size_t)b * L_ + lBase + 16 * j + q) * H_ + 8 * g2;

    v8f acc[4][2] = {};

    for (int uB = 0; uB < H_; uB += 32) {
        // Unconditional: speculative prefetch past the buffer end is silently
        // dropped (ISA 10.5). Locality 3 -> WGP-scope (fill near caches).
        __builtin_prefetch(bCol[0] + uB + 64, 0, 3);
        __builtin_prefetch(bCol[1] + uB + 64, 0, 3);

        v16h a[4], bf[2];
        #pragma unroll
        for (int i = 0; i < 4; ++i) {
            const v8h lo = *reinterpret_cast<const v8h*>(aRow[i] + uB);
            const v8h hi = *reinterpret_cast<const v8h*>(aRow[i] + uB + 16);
            #pragma unroll
            for (int e = 0; e < 8; ++e) { a[i][e] = lo[e]; a[i][e + 8] = hi[e]; }
        }
        #pragma unroll
        for (int j = 0; j < 2; ++j) {
            const v8h lo = *reinterpret_cast<const v8h*>(bCol[j] + uB);
            const v8h hi = *reinterpret_cast<const v8h*>(bCol[j] + uB + 16);
            #pragma unroll
            for (int e = 0; e < 8; ++e) { bf[j][e] = lo[e]; bf[j][e + 8] = hi[e]; }
        }
        #pragma unroll
        for (int i = 0; i < 4; ++i)
            #pragma unroll
            for (int j = 0; j < 2; ++j)
                acc[i][j] = __builtin_amdgcn_wmma_f32_16x16x32_f16(
                    false, a[i], false, bf[j], (short)0, acc[i][j], false, false);
    }

    // Store 4x2 D tiles + bias: row v = vBase+16i+8*g2+r, col l = lBase+16j+q
    #pragma unroll
    for (int i = 0; i < 4; ++i) {
        const int vr = vBase + 16 * i + 8 * g2;
        float* o0 = Out + ((size_t)b * H_ + vr) * L_ + lBase + q;
        #pragma unroll
        for (int r = 0; r < 8; ++r) {
            const float bias = out_b[vr + r];
            o0[(size_t)r * L_ + 0]  = acc[i][0][r] + bias;
            o0[(size_t)r * L_ + 16] = acc[i][1][r] + bias;
        }
    }
}

extern "C" void kernel_launch(void* const* d_in, const int* in_sizes, int n_in,
                              void* d_out, int out_size, void* d_ws, size_t ws_size,
                              hipStream_t stream) {
    (void)in_sizes; (void)n_in; (void)out_size; (void)ws_size;
    const float* u      = (const float*)d_in[0];
    const float* log_dt = (const float*)d_in[1];
    const float* Lambda = (const float*)d_in[2];
    const float* W      = (const float*)d_in[3];
    const float* D      = (const float*)d_in[4];
    const float* out_w  = (const float*)d_in[5];
    const float* out_b  = (const float*)d_in[6];
    float* Out = (float*)d_out;

    // Workspace: gT[b][l][u] f16 (25.2 MB, L2-resident) + Wh f16 (1.2 MB)
    _Float16* gT = (_Float16*)d_ws;
    _Float16* Wh = gT + (size_t)B_ * L_ * H_;

    // Pass 0: weight conversion, 768*768/1024 blocks
    convert_w_kernel<<<(H_ * H_) / 1024, 256, 0, stream>>>(out_w, Wh);

    // Pass 1: B*H = 6144 waves, 8 waves/block
    dss_scan_kernel<<<(B_ * H_) / 8, 256, 0, stream>>>(u, log_dt, Lambda, W, D, gT);

    // Pass 2: one wave per 64x32 macro-tile: 12 * 64 * 8 = 6144 waves
    const int totalWaves = (H_ / 64) * (L_ / 32) * B_;
    dss_out_gemm_kernel<<<totalWaves / 8, 256, 0, stream>>>(gT, Wh, out_b, Out);
}
